// LSTMModel_36807869727091
// MI455X (gfx1250) — matrix operand order repaired and checked by reading, not compiled, not run on hardware
//
#include <hip/hip_runtime.h>
#include <hip/hip_bf16.h>

typedef __attribute__((ext_vector_type(16))) _Float16 v16h;
typedef __attribute__((ext_vector_type(8)))  _Float16 v8h;
typedef __attribute__((ext_vector_type(8)))  float    v8f;

#define BB     4096
#define TT     1024
#define DIN    2
#define HH     16
#define NTILES (BB / 16)   // 256 batch tiles -> 256 blocks

__device__ __forceinline__ float act_tanh(float x) {
#if __has_builtin(__builtin_amdgcn_tanhf)
  return __builtin_amdgcn_tanhf(x);          // native v_tanh_f32 on gfx1250
#elif __has_builtin(__builtin_amdgcn_tanh_f32)
  return __builtin_amdgcn_tanh_f32(x);
#else
  float e = __expf(2.0f * x);
  return 1.0f - 2.0f / (e + 1.0f);
#endif
}

// One workgroup = 3 waves = one 16-row batch tile; wave l owns LSTM layer l.
// Transposed formulation: D'[gate_rows x batch] = W'[gate_rows x K] * Z[K x batch],
// with W' as four constant A-fragments and Z = [input16 ; h_own16] as B operand.
// D' comes out batch-per-lane == B-operand layout for the next step, so the
// recurrence never leaves registers (cvt_pk + lane16 swap + dword selects).
__global__ __launch_bounds__(96)
void lstm3_pipe_wmma_t(const float* __restrict__ x,
                       const float* __restrict__ h0,
                       const float* __restrict__ c0,
                       const float* __restrict__ Wih0, const float* __restrict__ Whh0,
                       const float* __restrict__ bih0, const float* __restrict__ bhh0,
                       const float* __restrict__ Wih1, const float* __restrict__ Whh1,
                       const float* __restrict__ bih1, const float* __restrict__ bhh1,
                       const float* __restrict__ Wih2, const float* __restrict__ Whh2,
                       const float* __restrict__ bih2, const float* __restrict__ bhh2,
                       const float* __restrict__ Wfc,  const float* __restrict__ bfc,
                       float* __restrict__ out)
{
  // publish buffers: [layer][parity][batch-col 0..15][16 halves] (32B per col)
  __shared__ __align__(32) _Float16 hbuf[3][2][16 * 16];

  const int lane = threadIdx.x & 31;
  const int L    = threadIdx.x >> 5;          // wave id == layer id (0..2)
  const int base = blockIdx.x * 16;           // batch base row of this tile
  const int nrow = lane & 15;                 // batch col / gate-row-in-tile
  const int khi  = lane >> 4;                 // lane-half selector
  const int kb8  = khi * 8;

  // Per-layer weight pointers (wave-uniform branches).
  const float *Wih, *Whh, *bih, *bhh;
  int din;
  if (L == 0)      { Wih = Wih0; Whh = Whh0; bih = bih0; bhh = bhh0; din = DIN; }
  else if (L == 1) { Wih = Wih1; Whh = Whh1; bih = bih1; bhh = bhh1; din = HH;  }
  else             { Wih = Wih2; Whh = Whh2; bih = bih2; bhh = bhh2; din = HH;  }

  // ---- Constant A-fragments: A_g[16x32] = W' rows 16g..16g+15, where
  // W'[f][k<16] = Wih[f][k] (zero-padded past din), W'[f][k>=16] = Whh[f][k-16].
  // Gates i,f,o (g!=2) pre-scaled by 0.5 so sigmoid = fma(tanh, 0.5, 0.5).
  // A 16x32 layout: lane row m = lane&15; half j -> K = khi*16 + j.
  // C-operand bias fragments: per-VGPR-slot gate row 16g+kb8+j, constant/lane.
  v16h Af[4];
  v8f  Cb[4];
#pragma unroll
  for (int g = 0; g < 4; ++g) {
    const float sc = (g == 2) ? 1.0f : 0.5f;
    const int f = g * 16 + nrow;              // global gate feature row
    v16h af;
#pragma unroll
    for (int j = 0; j < 16; ++j) {
      const int k = khi * 16 + j;
      float w = 0.0f;
      if (k < 16) {
        if (k < din) w = Wih[f * din + k];
      } else {
        w = Whh[f * HH + (k - 16)];
      }
      af[j] = (_Float16)(w * sc);
    }
    Af[g] = af;
    v8f cb;
#pragma unroll
    for (int j = 0; j < 8; ++j) {
      const int fr = g * 16 + kb8 + j;
      cb[j] = sc * (bih[fr] + bhh[fr]);
    }
    Cb[g] = cb;
  }

  // ---- State, batch-major: lane = batch col nrow.
  // c: VGPR j = hidden unit kb8+j.
  // own = h units kb8..kb8+7 (this lane half), par = the other 8 units.
  union H8  { v8h  h; int i[4]; };
  union Vz  { v16h h; int i[8]; };
  H8 own, par;
  v8f cst;
  {
    const size_t rb = ((size_t)L * BB + base + nrow) * HH;
#pragma unroll
    for (int j = 0; j < 8; ++j) {
      cst[j]   = c0[rb + kb8 + j];
      own.h[j] = (_Float16)h0[rb + kb8 + j];
      par.h[j] = (_Float16)h0[rb + (8 - kb8) + j];
    }
  }

  const float* xlane = x + (size_t)(base + nrow) * TT * DIN;  // wave 0 only
  v8f h2last = {};

  for (int s = 0; s < TT + 2; ++s) {
    const bool active = (s >= L) && (s < L + TT);
    if (active) {
      const int t = s - L;

      // ---- input16: layer input rows K=0..15 (consumed by lanes 0-15 of Z)
      Vz in;
      in.h = (v16h){};
      if (L == 0) {
        in.h[0] = (_Float16)xlane[t * DIN + 0];
        in.h[1] = (_Float16)xlane[t * DIN + 1];
        if (((t & 31) == 0) && (t + 64 < TT))
          __builtin_prefetch(xlane + (size_t)(t + 64) * DIN, 0, 0);
      } else {
        // h_{l-1}(t): published by upstream wave last step (other parity slot)
        in.h = *(const v16h*)(&hbuf[L - 1][(s - 1) & 1][nrow * 16]);
      }

      // ---- Z (B operand 32x16): lanes 0-15 hold K=0..15 (input16),
      // lanes 16-31 hold K=16..31 (own h = [par|own] = units 0..15 in order).
      // Dword-granular selects: 8 x v_cndmask_b32.
      Vz z;
#pragma unroll
      for (int d = 0; d < 4; ++d) {
        z.i[d]     = khi ? par.i[d] : in.i[d];
        z.i[d + 4] = khi ? own.i[d] : in.i[d + 4];
      }

      // ---- gates (bias pre-loaded via C operand; distinct accumulators)
      v8f gi = __builtin_amdgcn_wmma_f32_16x16x32_f16(false, Af[0], false, z.h, (short)0, Cb[0], false, false);
      v8f gf = __builtin_amdgcn_wmma_f32_16x16x32_f16(false, Af[1], false, z.h, (short)0, Cb[1], false, false);
      v8f gg = __builtin_amdgcn_wmma_f32_16x16x32_f16(false, Af[2], false, z.h, (short)0, Cb[2], false, false);
      v8f go = __builtin_amdgcn_wmma_f32_16x16x32_f16(false, Af[3], false, z.h, (short)0, Cb[3], false, false);

      // ---- elementwise (i,f,o inputs pre-scaled by 0.5 in weights/bias)
      v8f hnew;
#pragma unroll
      for (int j = 0; j < 8; ++j) {
        float ig = __builtin_fmaf(act_tanh(gi[j]), 0.5f, 0.5f);
        float fg = __builtin_fmaf(act_tanh(gf[j]), 0.5f, 0.5f);
        float cg = act_tanh(gg[j]);
        float og = __builtin_fmaf(act_tanh(go[j]), 0.5f, 0.5f);
        float cc = fg * cst[j] + ig * cg;
        cst[j] = cc;
        hnew[j] = og * act_tanh(cc);
      }
      if (L == 2) h2last = hnew;

      // ---- refresh own/par: pack to f16, swap halves with lane^16
#pragma unroll
      for (int j = 0; j < 8; ++j) own.h[j] = (_Float16)hnew[j];
#pragma unroll
      for (int d = 0; d < 4; ++d) par.i[d] = __shfl_xor(own.i[d], 16, 32);

      // ---- publish h(t) for downstream wave: lanes 0-15 hold [own|par] =
      // units 0..15 in order; contiguous 32B per batch col, zero selects.
      if (khi == 0) {
        Vz pub;
#pragma unroll
        for (int d = 0; d < 4; ++d) { pub.i[d] = own.i[d]; pub.i[d + 4] = par.i[d]; }
        *(v16h*)(&hbuf[L][s & 1][nrow * 16]) = pub.h;
      }
    }
    __syncthreads();   // step handoff between the 3 layer waves
  }

  // ---- FC head (wave 2): out[b] = sum_u h2[u][b] * Wfc[u] + bfc
  if (L == 2) {
    float p = 0.0f;
#pragma unroll
    for (int j = 0; j < 8; ++j)
      p = __builtin_fmaf(h2last[j], Wfc[kb8 + j], p);
    p += __shfl_xor(p, 16, 32);
    if (khi == 0) out[base + nrow] = p + bfc[0];
  }
}

extern "C" void kernel_launch(void* const* d_in, const int* in_sizes, int n_in,
                              void* d_out, int out_size, void* d_ws, size_t ws_size,
                              hipStream_t stream) {
  (void)in_sizes; (void)n_in; (void)d_ws; (void)ws_size; (void)out_size;
  const float* x    = (const float*)d_in[0];
  const float* h0   = (const float*)d_in[1];
  const float* c0   = (const float*)d_in[2];
  const float* Wih0 = (const float*)d_in[3];
  const float* Whh0 = (const float*)d_in[4];
  const float* bih0 = (const float*)d_in[5];
  const float* bhh0 = (const float*)d_in[6];
  const float* Wih1 = (const float*)d_in[7];
  const float* Whh1 = (const float*)d_in[8];
  const float* bih1 = (const float*)d_in[9];
  const float* bhh1 = (const float*)d_in[10];
  const float* Wih2 = (const float*)d_in[11];
  const float* Whh2 = (const float*)d_in[12];
  const float* bih2 = (const float*)d_in[13];
  const float* bhh2 = (const float*)d_in[14];
  const float* Wfc  = (const float*)d_in[15];
  const float* bfc  = (const float*)d_in[16];

  dim3 grid(NTILES);   // 256 blocks: one 16-row batch tile each
  dim3 block(96);      // 3 waves: one LSTM layer each (software pipeline)
  lstm3_pipe_wmma_t<<<grid, block, 0, stream>>>(x, h0, c0,
      Wih0, Whh0, bih0, bhh0, Wih1, Whh1, bih1, bhh1,
      Wih2, Whh2, bih2, bhh2, Wfc, bfc, (float*)d_out);
}
